// GearNet_30588757082312
// MI455X (gfx1250) — matrix operand order repaired and checked by reading, not compiled, not run on hardware
//
#include <hip/hip_runtime.h>

#define N_NODES 50000
#define N_EDGES 800000
#define D       256
#define ED      128
#define RREL    7
#define EPS_BN  1e-5f
#define SLOPE   0.1f

typedef __attribute__((ext_vector_type(16))) __bf16 v16bf;
typedef __attribute__((ext_vector_type(8)))  float  v8f;

// LeakyReLU with positive slope < 1:  lrelu(x) == max(x, SLOPE*x)
__device__ __forceinline__ float lrelu(float x) { return fmaxf(x, SLOPE * x); }

// Build a 16x32 bf16 A-fragment with fused BN + LeakyReLU prologue.
__device__ __forceinline__ v16bf build_a(const float* __restrict__ xr,
                                         const float* __restrict__ s1,
                                         const float* __restrict__ t1, int kb) {
  v16bf a;
#pragma unroll
  for (int j = 0; j < 8; ++j) {
    float x = xr[kb + j] * s1[kb + j] + t1[kb + j];
    a[j] = (__bf16)lrelu(x);
  }
#pragma unroll
  for (int j = 0; j < 8; ++j) {
    int k = kb + 16 + j;
    float x = xr[k] * s1[k] + t1[k];
    a[8 + j] = (__bf16)lrelu(x);
  }
  return a;
}

#define LDFRAG(sb, ntc, boff) (*(const v16bf*)((sb) + (size_t)(ntc) * 1024 + (boff)))

#if defined(__has_builtin)
#if __has_builtin(__builtin_amdgcn_sched_group_barrier)
#define HAVE_SGB 1
#endif
#endif

// Force the ds_read / WMMA software pipeline the pressure-aware scheduler
// refuses to build on its own. Masks: 0x020=VMEM read, 0x100=DS read,
// 0x008=MFMA/WMMA, 0x200=DS write.
__device__ __forceinline__ void sched_wmma_pipeline(bool with_stores, int nglobal) {
#ifdef HAVE_SGB
  __builtin_amdgcn_sched_group_barrier(0x020, 8, 0);   // global reads issue early
  __builtin_amdgcn_sched_group_barrier(0x100, 8, 0);   // 4-fragment DS preload
#pragma unroll
  for (int i = 0; i < 12; ++i) {
    __builtin_amdgcn_sched_group_barrier(0x008, 1, 0); // 1 WMMA
    __builtin_amdgcn_sched_group_barrier(0x100, 2, 0); // next fragment's 2 ds_reads
  }
  __builtin_amdgcn_sched_group_barrier(0x008, 4, 0);   // tail WMMAs
  if (with_stores)
    __builtin_amdgcn_sched_group_barrier(0x200, 4, 0); // LDS staging stores last
#endif
  (void)with_stores; (void)nglobal;
}

// ---------------------------------------------------------------------------
// Fold BatchNorm (g,b,m,v) into per-channel scale s and shift t.
// ---------------------------------------------------------------------------
__global__ void prep_bn_kernel(const float* __restrict__ g, const float* __restrict__ b,
                               const float* __restrict__ m, const float* __restrict__ v,
                               float* __restrict__ s, float* __restrict__ t, int dim) {
  int i = blockIdx.x * blockDim.x + threadIdx.x;
  if (i < dim) {
    float sc = g[i] * rsqrtf(v[i] + EPS_BN);
    s[i] = sc;
    t[i] = b[i] - m[i] * sc;
  }
}

// ---------------------------------------------------------------------------
// Pack fp32 weights [K, NOUT] (row-major) into bf16 WMMA B-fragments.
//   n = lane & 15, h = lane >> 4: K = k0 + 8h + j (j<8), K = k0 + 16 + 8h + (j-8) (j>=8)
//   Wpk[ ((kstep*NT + nt)*32 + lane)*16 + j ]
// ---------------------------------------------------------------------------
__global__ void pack_w_kernel(const float* __restrict__ W, __bf16* __restrict__ Wpk,
                              int K, int NOUT) {
  int idx = blockIdx.x * blockDim.x + threadIdx.x;
  if (idx >= K * NOUT) return;
  int j    = idx & 15;
  int lane = (idx >> 4) & 31;
  int frag = idx >> 9;
  int NT   = NOUT >> 4;
  int nt   = frag % NT;
  int ks   = frag / NT;
  int k = ks * 32 + ((lane >> 4) << 3) + (j & 7) + ((j >> 3) << 4);
  int n = nt * 16 + (lane & 15);
  Wpk[idx] = (__bf16)W[k * NOUT + n];
}

// ---------------------------------------------------------------------------
// Fused  Y = epilogue( lrelu(X*s1+t1) @ W ), NOUT = 256.
// One wave: 16 rows x 256 cols (16 accumulators); 8 waves/block.
// B tiles double-buffered through LDS; B fragments pipelined 4-deep in
// registers with an explicit sched_group_barrier pipeline.
// ---------------------------------------------------------------------------
__global__ __launch_bounds__(256) void gemm_bn_wmma(
    const float* __restrict__ X, int M, int K,
    const float* __restrict__ s1, const float* __restrict__ t1,
    const __bf16* __restrict__ Wpk,
    const float* __restrict__ bias,
    const float* __restrict__ s2, const float* __restrict__ t2,
    const float* __restrict__ resid,
    float* __restrict__ Y) {
  __shared__ __align__(16) unsigned char smem[2 * 16384];  // double-buffered B tile
  const int tid    = threadIdx.x;
  const int waveId = tid >> 5;
  const int lane   = tid & 31;
  const int half   = lane >> 4;
  const int ln     = lane & 15;
  const int boff   = lane * 32;

  int mt = blockIdx.x * 8 + waveId;
  const int  nmt    = M >> 4;
  const bool active = (mt < nmt);           // wave-uniform
  if (!active) mt = 0;                      // clamp: still participates in barriers
  const float* xr = X + (size_t)(mt * 16 + ln) * K;

  v8f acc[16];
#pragma unroll
  for (int i = 0; i < 16; ++i) acc[i] = (v8f){};

  const int nks = K >> 5;

  // Preload B tile 0 into LDS buffer 0 (1024 uint4 per tile, 4 per thread).
  {
    const uint4* g  = (const uint4*)Wpk;
    uint4*       sb = (uint4*)smem;
#pragma unroll
    for (int i = 0; i < 4; ++i) sb[tid + 256 * i] = g[tid + 256 * i];
  }

  for (int ks = 0; ks < nks; ++ks) {
    __syncthreads();  // current buffer ready; previous reads of next buffer done
    // Branch-free staging: last iteration clamps to the current tile and
    // rewrites identical bytes into the buffer being read (value-benign),
    // keeping this body one scheduling region.
    const int ksn = (ks + 1 < nks) ? (ks + 1) : ks;
    const uint4* g = (const uint4*)(Wpk + (size_t)ksn * 8192);
    uint4 r0 = g[tid];
    uint4 r1 = g[tid + 256];
    uint4 r2 = g[tid + 512];
    uint4 r3 = g[tid + 768];

    const unsigned char* sb = smem + (size_t)(ks & 1) * 16384;
    v16bf b0 = LDFRAG(sb, 0, boff);
    v16bf b1 = LDFRAG(sb, 1, boff);
    v16bf b2 = LDFRAG(sb, 2, boff);
    v16bf b3 = LDFRAG(sb, 3, boff);
    const v16bf a = build_a(xr, s1, t1, ks * 32 + half * 8);

#pragma unroll
    for (int nt = 0; nt < 16; nt += 4) {
      acc[nt + 0] = __builtin_amdgcn_wmma_f32_16x16x32_bf16(false, a, false, b0, (short)0,
                                                            acc[nt + 0], false, false);
      if (nt + 4 < 16) b0 = LDFRAG(sb, nt + 4, boff);
      acc[nt + 1] = __builtin_amdgcn_wmma_f32_16x16x32_bf16(false, a, false, b1, (short)0,
                                                            acc[nt + 1], false, false);
      if (nt + 5 < 16) b1 = LDFRAG(sb, nt + 5, boff);
      acc[nt + 2] = __builtin_amdgcn_wmma_f32_16x16x32_bf16(false, a, false, b2, (short)0,
                                                            acc[nt + 2], false, false);
      if (nt + 6 < 16) b2 = LDFRAG(sb, nt + 6, boff);
      acc[nt + 3] = __builtin_amdgcn_wmma_f32_16x16x32_bf16(false, a, false, b3, (short)0,
                                                            acc[nt + 3], false, false);
      if (nt + 7 < 16) b3 = LDFRAG(sb, nt + 7, boff);
    }

    uint4* d = (uint4*)(smem + (size_t)(ksn & 1) * 16384);
    d[tid]       = r0;
    d[tid + 256] = r1;
    d[tid + 512] = r2;
    d[tid + 768] = r3;

    sched_wmma_pipeline(true, 8);
  }

  if (active) {
#pragma unroll
    for (int nt = 0; nt < 16; ++nt) {
      const int n  = nt * 16 + ln;
      const float bv = bias ? bias[n] : 0.f;
      float sv = 1.f, tv = 0.f;
      const bool post = (s2 != nullptr);
      if (post) { sv = s2[n]; tv = t2[n]; }
#pragma unroll
      for (int i = 0; i < 8; ++i) {
        long m = (long)mt * 16 + i + half * 8;
        float val = acc[nt][i] + bv;
        if (post) val = lrelu(val * sv + tv);
        if (resid) val += resid[m * D + n];
        Y[m * D + n] = val;
      }
    }
  }
}

// ---------------------------------------------------------------------------
// Edge message kernel: one wave = 16 edges x 256 cols; 8 waves/block.
// Whole packed edge weight (64KB) staged in LDS once; 64 WMMAs per wave,
// B fragments pipelined 4-deep with sched_group_barrier.
// ---------------------------------------------------------------------------
__global__ __launch_bounds__(256) void edge_msg_wmma(
    const float* __restrict__ HE,
    const int* __restrict__ nin, const int* __restrict__ nout, const int* __restrict__ rel,
    const float* __restrict__ s1, const float* __restrict__ t1,
    const __bf16* __restrict__ Wpk,
    const float* __restrict__ bias,
    const float* __restrict__ s2, const float* __restrict__ t2,
    const float* __restrict__ hv,
    float* __restrict__ upd) {
  __shared__ __align__(16) unsigned char smem[ED * D * 2];  // 64KB: whole W
  const int tid = threadIdx.x;
  {
    const uint4* g  = (const uint4*)Wpk;
    uint4*       sb = (uint4*)smem;
#pragma unroll
    for (int i = 0; i < 16; ++i) sb[tid + 256 * i] = g[tid + 256 * i];
  }
  __syncthreads();

  const int waveId = tid >> 5;
  const int lane   = tid & 31;
  const int half   = lane >> 4;
  const int ln     = lane & 15;
  const int boff   = lane * 32;
  const int e0     = blockIdx.x * 128 + waveId * 16;
  const float* xr  = HE + (size_t)(e0 + ln) * ED;

  v8f acc[16];
#pragma unroll
  for (int i = 0; i < 16; ++i) acc[i] = (v8f){};

#pragma unroll
  for (int ks = 0; ks < (ED >> 5); ++ks) {
    const unsigned char* sb = smem + (size_t)ks * 16384;
    v16bf b0 = LDFRAG(sb, 0, boff);
    v16bf b1 = LDFRAG(sb, 1, boff);
    v16bf b2 = LDFRAG(sb, 2, boff);
    v16bf b3 = LDFRAG(sb, 3, boff);
    const v16bf a = build_a(xr, s1, t1, ks * 32 + half * 8);
#pragma unroll
    for (int nt = 0; nt < 16; nt += 4) {
      acc[nt + 0] = __builtin_amdgcn_wmma_f32_16x16x32_bf16(false, a, false, b0, (short)0,
                                                            acc[nt + 0], false, false);
      if (nt + 4 < 16) b0 = LDFRAG(sb, nt + 4, boff);
      acc[nt + 1] = __builtin_amdgcn_wmma_f32_16x16x32_bf16(false, a, false, b1, (short)0,
                                                            acc[nt + 1], false, false);
      if (nt + 5 < 16) b1 = LDFRAG(sb, nt + 5, boff);
      acc[nt + 2] = __builtin_amdgcn_wmma_f32_16x16x32_bf16(false, a, false, b2, (short)0,
                                                            acc[nt + 2], false, false);
      if (nt + 6 < 16) b2 = LDFRAG(sb, nt + 6, boff);
      acc[nt + 3] = __builtin_amdgcn_wmma_f32_16x16x32_bf16(false, a, false, b3, (short)0,
                                                            acc[nt + 3], false, false);
      if (nt + 7 < 16) b3 = LDFRAG(sb, nt + 7, boff);
    }
    sched_wmma_pipeline(false, 4);
  }

#pragma unroll
  for (int i = 0; i < 8; ++i) {
    const int e = e0 + i + half * 8;
    const size_t nirow = (size_t)nin[e] * D;
    const size_t seg   = ((size_t)nout[e] * RREL + (size_t)rel[e]) * D;
#pragma unroll
    for (int nt = 0; nt < 16; ++nt) {
      const int n = nt * 16 + ln;
      float val = lrelu((acc[nt][i] + bias[n]) * s2[n] + t2[n]);
      val += hv[nirow + n];
      __hip_atomic_fetch_add(&upd[seg + n], val, __ATOMIC_RELAXED, __HIP_MEMORY_SCOPE_AGENT);
    }
  }
}

// ---------------------------------------------------------------------------
// Host side
// ---------------------------------------------------------------------------
extern "C" void kernel_launch(void* const* d_in, const int* in_sizes, int n_in,
                              void* d_out, int out_size, void* d_ws, size_t ws_size,
                              hipStream_t stream) {
  (void)in_sizes; (void)n_in; (void)out_size; (void)ws_size;
  const float* h_v  = (const float*)d_in[0];
  const int*   eidx = (const int*)d_in[1];
  const float* h_e  = (const float*)d_in[2];
  const int* nin  = eidx;
  const int* nout = eidx + N_EDGES;
  const int* rel  = eidx + 2 * N_EDGES;

  // params flattened pytree-style (dict keys sorted): leaves start at d_in[3].
  auto LP = [&](int l, int o) -> const float* { return (const float*)d_in[3 + l * 30 + o]; };
  auto MP = [&](int o)        -> const float* { return (const float*)d_in[3 + 90 + o]; };

  // ---- carve workspace ----
  size_t off = 0;
  auto alloc = [&](size_t bytes) -> void* {
    void* p = (char*)d_ws + off;
    off += (bytes + 255) & ~(size_t)255;
    return p;
  };
  float* bufA = (float*)alloc((size_t)N_NODES * D * 4);
  float* bufB = (float*)alloc((size_t)N_NODES * D * 4);
  float* hv   = (float*)alloc((size_t)N_NODES * D * 4);
  float* upd  = (float*)alloc((size_t)N_NODES * RREL * D * 4);

  __bf16 *WinP[3], *WedgeP[3], *WlinP[3], *WoutP[3], *WmlpP;
  float *inS1[3], *inT1[3], *inS2[3], *inT2[3];
  float *eS1[3], *eT1[3], *eS2[3], *eT2[3];
  float *lS[3], *lT[3], *oS[3], *oT[3];
  for (int l = 0; l < 3; ++l) {
    WinP[l]   = (__bf16*)alloc((size_t)D * D * 2);
    WedgeP[l] = (__bf16*)alloc((size_t)ED * D * 2);
    WlinP[l]  = (__bf16*)alloc((size_t)RREL * D * D * 2);
    WoutP[l]  = (__bf16*)alloc((size_t)D * D * 2);
    inS1[l] = (float*)alloc(D * 4);      inT1[l] = (float*)alloc(D * 4);
    inS2[l] = (float*)alloc(D * 4);      inT2[l] = (float*)alloc(D * 4);
    eS1[l]  = (float*)alloc(ED * 4);     eT1[l]  = (float*)alloc(ED * 4);
    eS2[l]  = (float*)alloc(D * 4);      eT2[l]  = (float*)alloc(D * 4);
    lS[l]   = (float*)alloc(RREL * D * 4); lT[l] = (float*)alloc(RREL * D * 4);
    oS[l]   = (float*)alloc(D * 4);      oT[l]  = (float*)alloc(D * 4);
  }
  WmlpP = (__bf16*)alloc((size_t)D * D * 2);
  float* mS1 = (float*)alloc(D * 4); float* mT1 = (float*)alloc(D * 4);
  float* mS2 = (float*)alloc(D * 4); float* mT2 = (float*)alloc(D * 4);

  // ---- prep: fold BN params, pack weights to bf16 fragments ----
  auto prep = [&](const float* b, const float* g, const float* m, const float* v,
                  float* s, float* t, int dim) {
    prep_bn_kernel<<<(dim + 255) / 256, 256, 0, stream>>>(g, b, m, v, s, t, dim);
  };
  auto pack = [&](const float* W, __bf16* Wpk, int K) {
    int tot = K * D;
    pack_w_kernel<<<(tot + 255) / 256, 256, 0, stream>>>(W, Wpk, K, D);
  };
  for (int l = 0; l < 3; ++l) {
    prep(LP(l, 12), LP(l, 13), LP(l, 14), LP(l, 15), inS1[l], inT1[l], D);   // input.bn1
    prep(LP(l, 16), LP(l, 17), LP(l, 18), LP(l, 19), inS2[l], inT2[l], D);   // input.bn2
    prep(LP(l, 2),  LP(l, 3),  LP(l, 4),  LP(l, 5),  eS1[l],  eT1[l],  ED);  // edge.bn1
    prep(LP(l, 6),  LP(l, 7),  LP(l, 8),  LP(l, 9),  eS2[l],  eT2[l],  D);   // edge.bn2
    prep(LP(l, 21), LP(l, 22), LP(l, 23), LP(l, 24), lS[l],   lT[l],   RREL * D); // linear.bn
    prep(LP(l, 26), LP(l, 27), LP(l, 28), LP(l, 29), oS[l],   oT[l],   D);   // output.bn
    pack(LP(l, 10), WinP[l],   D);
    pack(LP(l, 0),  WedgeP[l], ED);
    pack(LP(l, 20), WlinP[l],  RREL * D);
    pack(LP(l, 25), WoutP[l],  D);
  }
  prep(MP(2), MP(3), MP(4), MP(5), mS1, mT1, D);   // mlp.bn1
  prep(MP(6), MP(7), MP(8), MP(9), mS2, mT2, D);   // mlp.bn2
  pack(MP(0), WmlpP, D);

  // ---- main layers ----
  const int node_mtiles = N_NODES / 16;               // 3125
  const int node_blocks = (node_mtiles + 7) / 8;      // 391 (last block partial)
  const int edge_blocks = N_EDGES / 128;              // 6250

  const float* hcur = h_v;
  float* hbuf[2] = {bufA, bufB};
  for (int l = 0; l < 3; ++l) {
    // hv = input-MLP(h)
    gemm_bn_wmma<<<node_blocks, 256, 0, stream>>>(
        hcur, N_NODES, D, inS1[l], inT1[l], WinP[l],
        LP(l, 11), inS2[l], inT2[l], nullptr, hv);

    // upd = segment_sum(hv[node_in] + edge-MLP(h_e), node_out*R + rel)
    hipMemsetAsync(upd, 0, (size_t)N_NODES * RREL * D * 4, stream);
    edge_msg_wmma<<<edge_blocks, 256, 0, stream>>>(
        h_e, nin, nout, rel, eS1[l], eT1[l], WedgeP[l],
        LP(l, 1), eS2[l], eT2[l], hv, upd);

    // tmp (reuse hv) = linear(upd)  [K = R*D = 1792]
    gemm_bn_wmma<<<node_blocks, 256, 0, stream>>>(
        upd, N_NODES, RREL * D, lS[l], lT[l], WlinP[l],
        nullptr, nullptr, nullptr, nullptr, hv);

    // h_next = output(tmp) + h
    float* hnext = hbuf[l & 1];
    gemm_bn_wmma<<<node_blocks, 256, 0, stream>>>(
        hv, N_NODES, D, oS[l], oT[l], WoutP[l],
        nullptr, nullptr, nullptr, hcur, hnext);
    hcur = hnext;
  }

  // final MLP -> d_out
  gemm_bn_wmma<<<node_blocks, 256, 0, stream>>>(
      hcur, N_NODES, D, mS1, mT1, WmlpP,
      MP(1), mS2, mT2, nullptr, (float*)d_out);
}
